// CausalAttention_22136261444412
// MI455X (gfx1250) — matrix operand order repaired and checked by reading, not compiled, not run on hardware
//
#include <hip/hip_runtime.h>

typedef __bf16 bf16;
typedef __attribute__((ext_vector_type(16))) __bf16 v16bf;
typedef __attribute__((ext_vector_type(8)))  __bf16 v8bf;
typedef __attribute__((ext_vector_type(8)))  float  v8f;
typedef __attribute__((ext_vector_type(4)))  int    v4i;

// B=4, T=2048, C=1024, NH=16, HD=64
#define BT   8192      // B*T rows
#define CDIM 1024
#define C3   3072
#define TSEQ 2048
#define NHEAD 16

#if __has_builtin(__builtin_amdgcn_global_load_async_to_lds_b128)
#define HAVE_ASYNC_LDS 1
typedef __attribute__((address_space(1))) v4i* as1_v4i;
typedef __attribute__((address_space(3))) v4i* as3_v4i;
#else
#define HAVE_ASYNC_LDS 0
#endif

__device__ __forceinline__ bf16 f2bf(float f) {
  union { float f; unsigned u; } v; v.f = f;
  unsigned r = v.u + 0x7fffu + ((v.u >> 16) & 1u);   // round-to-nearest-even
  union { unsigned short s; bf16 b; } o; o.s = (unsigned short)(r >> 16);
  return o.b;
}

// 16-byte global -> LDS copy. Prefer CDNA5 async-to-LDS DMA (ASYNCcnt, no VGPR
// round trip); fall back to register staging if the builtin is unavailable.
__device__ __forceinline__ void cp16(bf16* l, const bf16* g) {
#if HAVE_ASYNC_LDS
  __builtin_amdgcn_global_load_async_to_lds_b128(
      (as1_v4i)(void*)g, (as3_v4i)(void*)l, 0, 0);
#else
  *(v8bf*)l = *(const v8bf*)g;
#endif
}
__device__ __forceinline__ void async_fence() {
#if HAVE_ASYNC_LDS
#if __has_builtin(__builtin_amdgcn_s_wait_asynccnt)
  __builtin_amdgcn_s_wait_asynccnt(0);
#else
  asm volatile("s_wait_asynccnt 0x0" ::: "memory");
#endif
#endif
}

// ---- WMMA 16x16x32 bf16 fragment loaders (ISA 7.12.2 layouts) ----
// A (16x32, MxK): lane L = row M=L%16. Lanes 0-15 hold K {0..7,16..23},
// lanes 16-31 hold K {8..15,24..31}  -> two ds_load_b128 at +0/+16 halves.
__device__ __forceinline__ v16bf frag_ld_a(const bf16* rowp, int lane, int k0) {
  const bf16* q = rowp + k0 + ((lane & 16) >> 1);
  v16bf f;
  *(v8bf*)&f       = *(const v8bf*)q;
  *((v8bf*)&f + 1) = *(const v8bf*)(q + 16);
  return f;
}
// B (32x16, stored N-major Bt[n][k]): lane L = col N=L%16. Lanes 0-15 hold
// K {0..7,16..23}, lanes 16-31 hold K {8..15,24..31} -> matches A layout.
__device__ __forceinline__ v16bf frag_ld_b(const bf16* rowp, int lane, int k0) {
  const bf16* q = rowp + k0 + ((lane & 16) >> 1);
  v16bf f;
  *(v8bf*)&f       = *(const v8bf*)q;
  *((v8bf*)&f + 1) = *(const v8bf*)(q + 16);
  return f;
}

#define WMMA_BF16(A_, B_, C_) \
  __builtin_amdgcn_wmma_f32_16x16x32_bf16(false, (A_), false, (B_), (short)0, (C_), false, false)

// ---------------- pre-pass conversions ----------------
__global__ __launch_bounds__(256) void cvt_f32_bf16(const float* __restrict__ s,
                                                    bf16* __restrict__ d, int n4) {
  int i = blockIdx.x * 256 + threadIdx.x;
  if (i < n4) {
    float4 v = ((const float4*)s)[i];
    d[i * 4 + 0] = f2bf(v.x); d[i * 4 + 1] = f2bf(v.y);
    d[i * 4 + 2] = f2bf(v.z); d[i * 4 + 3] = f2bf(v.w);
  }
}

// src f32 [K x N] row-major  ->  dst bf16 [N x K] row-major (transposed)
__global__ __launch_bounds__(256) void cvt_transpose(const float* __restrict__ s,
                                                     bf16* __restrict__ d, int K, int N) {
  long long i = (long long)blockIdx.x * 256 + threadIdx.x;
  if (i < (long long)K * N) {
    int n = (int)(i / K), k = (int)(i - (long long)n * K);
    d[i] = f2bf(s[(size_t)k * N + n]);
  }
}

// ---------------- tiled WMMA GEMM ----------------
// C[M x N] = A[M x K] * Bt[N x K]^T (+ bias).  BM=128, BN=128, BK=64.
// 256 threads = 8 waves arranged 4(M) x 2(N), each wave owns 32x64
// (2x4 WMMA tiles: 6 fragment loads feed 8 WMMAs per K-step).
template <int OUT_BF16, int BIAS>
__global__ __launch_bounds__(256) void gemm_bf16_wmma(
    const bf16* __restrict__ A, const bf16* __restrict__ Bt,
    const float* __restrict__ bias, void* __restrict__ Cp,
    int M, int N, int K) {
  constexpr int BM = 128, BN = 128, BK = 64, ST = BK + 8;  // 72-half rows (144B)
  __shared__ __align__(16) bf16 sA[BM * ST];
  __shared__ __align__(16) bf16 sB[BN * ST];
  const int tid = threadIdx.x, lane = tid & 31, wave = tid >> 5;
  const int wm = (wave & 3) * 32, wn = (wave >> 2) * 64;
  const int m0 = blockIdx.y * BM, n0 = blockIdx.x * BN;
  const int ln = lane & 15, hf = (lane >> 4) << 3;
  v8f acc[2][4] = {};

  for (int k0 = 0; k0 < K; k0 += BK) {
#pragma unroll
    for (int i = 0; i < 4; ++i) {               // A tile: 1024 x 16B chunks
      int c = tid + 256 * i, r = c >> 3, col = (c & 7) * 8;
      cp16(&sA[r * ST + col], &A[(size_t)(m0 + r) * K + k0 + col]);
    }
#pragma unroll
    for (int i = 0; i < 4; ++i) {               // Bt tile: 1024 x 16B chunks
      int c = tid + 256 * i, r = c >> 3, col = (c & 7) * 8;
      cp16(&sB[r * ST + col], &Bt[(size_t)(n0 + r) * K + k0 + col]);
    }
    if (k0 + BK < K) {                          // hint next tiles toward L2
      __builtin_prefetch(&A[(size_t)(m0 + (tid & 127)) * K + k0 + BK], 0, 1);
      __builtin_prefetch(&Bt[(size_t)(n0 + (tid & 127)) * K + k0 + BK], 0, 1);
    }
    async_fence();
    __syncthreads();
#pragma unroll
    for (int ks = 0; ks < BK; ks += 32) {
      v16bf a0 = frag_ld_a(&sA[(wm + 0  + ln) * ST], lane, ks);
      v16bf a1 = frag_ld_a(&sA[(wm + 16 + ln) * ST], lane, ks);
      v16bf b[4];
#pragma unroll
      for (int j = 0; j < 4; ++j)
        b[j] = frag_ld_b(&sB[(wn + j * 16 + ln) * ST], lane, ks);
#pragma unroll
      for (int j = 0; j < 4; ++j) {
        acc[0][j] = WMMA_BF16(a0, b[j], acc[0][j]);
        acc[1][j] = WMMA_BF16(a1, b[j], acc[1][j]);
      }
    }
    __syncthreads();
  }
#pragma unroll
  for (int i = 0; i < 2; ++i)
#pragma unroll
    for (int j = 0; j < 4; ++j) {
      int col = n0 + wn + j * 16 + ln;
      float bs = BIAS ? bias[col] : 0.f;
#pragma unroll
      for (int r = 0; r < 8; ++r) {             // D layout: lane=col, VGPR r=row
        int row = m0 + wm + i * 16 + hf + r;
        float v = acc[i][j][r] + bs;
        if (OUT_BF16) ((bf16*)Cp)[(size_t)row * N + col] = f2bf(v);
        else          ((float*)Cp)[(size_t)row * N + col] = v;
      }
    }
}

// ---------------- flash attention ----------------
// grid = (T/64, B*NH); 128 threads = 4 waves, each wave owns 16 query rows.
__global__ __launch_bounds__(128) void attn_flash_wmma(const bf16* __restrict__ qkv,
                                                       bf16* __restrict__ ctx) {
  constexpr int ST = 72;
  __shared__ __align__(16) bf16 sQ[64 * ST];      // [query][hd]
  __shared__ __align__(16) bf16 sK[64 * ST];      // [key][hd]   (acts as Bt for QK^T)
  __shared__ __align__(16) bf16 sV[64 * ST];      // [hd][key]   (transposed -> Bt for PV)
  __shared__ __align__(16) bf16 sP[4][16 * ST];   // per-wave P strip [row][key]
  const int tid = threadIdx.x, lane = tid & 31, wave = tid >> 5;
  const int ln = lane & 15, hf = (lane >> 4) << 3;
  const int q0 = blockIdx.x * 64;
  const int b = blockIdx.y >> 4, h = blockIdx.y & 15;
  const size_t rs = C3;                            // qkv row stride (halves)
  const bf16* Qg = qkv + (size_t)b * TSEQ * rs + h * 64;
  const bf16* Kg = Qg + CDIM;
  const bf16* Vg = Qg + 2 * CDIM;

#pragma unroll
  for (int i = 0; i < 4; ++i) {                    // Q tile 64x64
    int c = tid + 128 * i, r = c >> 3, col = (c & 7) * 8;
    cp16(&sQ[r * ST + col], &Qg[(size_t)(q0 + r) * rs + col]);
  }
  const int qw = q0 + wave * 16;                   // this wave's query base
  float m_i[8], l_i[8];
  v8f o[4] = {};
#pragma unroll
  for (int r = 0; r < 8; ++r) { m_i[r] = -__builtin_inff(); l_i[r] = 0.f; }

  const int jmax = blockIdx.x;                     // causal: keys <= q0+63
  for (int j = 0; j <= jmax; ++j) {
    __syncthreads();
#pragma unroll
    for (int i = 0; i < 4; ++i) {                  // K tile 64x64
      int c = tid + 128 * i, r = c >> 3, col = (c & 7) * 8;
      cp16(&sK[r * ST + col], &Kg[(size_t)(j * 64 + r) * rs + col]);
    }
    {                                              // V tile, transposed on store
      int r = tid >> 1, cb = (tid & 1) * 32;
      v8bf tmp[4];
#pragma unroll
      for (int u = 0; u < 4; ++u)
        tmp[u] = *(const v8bf*)&Vg[(size_t)(j * 64 + r) * rs + cb + u * 8];
#pragma unroll
      for (int u = 0; u < 4; ++u)
#pragma unroll
        for (int e = 0; e < 8; ++e)
          sV[(cb + u * 8 + e) * ST + r] = tmp[u][e];
    }
    async_fence();
    __syncthreads();

    // S strip (16x64) = Q_strip @ K^T
    v8f s[4] = {};
#pragma unroll
    for (int ks = 0; ks < 64; ks += 32) {
      v16bf aq = frag_ld_a(&sQ[(wave * 16 + ln) * ST], lane, ks);
#pragma unroll
      for (int nt = 0; nt < 4; ++nt) {
        v16bf bk = frag_ld_b(&sK[(nt * 16 + ln) * ST], lane, ks);
        s[nt] = WMMA_BF16(aq, bk, s[nt]);
      }
    }
    // scale + causal mask (element: row qw+hf+r, col j*64+nt*16+ln)
#pragma unroll
    for (int nt = 0; nt < 4; ++nt) {
      int kcol = j * 64 + nt * 16 + ln;
#pragma unroll
      for (int r = 0; r < 8; ++r) {
        float v = s[nt][r] * 0.125f;
        if (kcol > qw + hf + r) v = -__builtin_inff();
        s[nt][r] = v;
      }
    }
    // online softmax: row stats via 16-lane half-wave reductions
    float mnew[8], alpha[8];
#pragma unroll
    for (int r = 0; r < 8; ++r) {
      float mx = fmaxf(fmaxf(s[0][r], s[1][r]), fmaxf(s[2][r], s[3][r]));
#pragma unroll
      for (int off = 8; off; off >>= 1) mx = fmaxf(mx, __shfl_xor(mx, off, 16));
      mnew[r] = fmaxf(m_i[r], mx);
      alpha[r] = __expf(m_i[r] - mnew[r]);
      m_i[r] = mnew[r];
    }
#pragma unroll
    for (int r = 0; r < 8; ++r) {
      float sum = 0.f;
#pragma unroll
      for (int nt = 0; nt < 4; ++nt) {
        float p = __expf(s[nt][r] - mnew[r]);
        s[nt][r] = p; sum += p;
      }
#pragma unroll
      for (int off = 8; off; off >>= 1) sum += __shfl_xor(sum, off, 16);
      l_i[r] = l_i[r] * alpha[r] + sum;
#pragma unroll
      for (int n = 0; n < 4; ++n) o[n][r] *= alpha[r];
    }
    // P -> per-wave LDS (C-layout -> row-major), then O += P @ V
    bf16* Pw = &sP[wave][0];
#pragma unroll
    for (int nt = 0; nt < 4; ++nt)
#pragma unroll
      for (int r = 0; r < 8; ++r)
        Pw[(hf + r) * ST + nt * 16 + ln] = f2bf(s[nt][r]);
    // DS ops are in-order within a wave: safe to read P strip back
#pragma unroll
    for (int ks = 0; ks < 64; ks += 32) {
      v16bf ap = frag_ld_a(&Pw[ln * ST], lane, ks);
#pragma unroll
      for (int n = 0; n < 4; ++n) {
        v16bf bv = frag_ld_b(&sV[(n * 16 + ln) * ST], lane, ks);
        o[n] = WMMA_BF16(ap, bv, o[n]);
      }
    }
  }
  // normalize + write ctx (bf16, [B*T, C] with head offset)
#pragma unroll
  for (int r = 0; r < 8; ++r) {
    float inv = 1.f / l_i[r];
    int row = qw + hf + r;
    bf16* dst = ctx + ((size_t)(b * TSEQ + row)) * CDIM + h * 64;
#pragma unroll
    for (int n = 0; n < 4; ++n) dst[n * 16 + ln] = f2bf(o[n][r] * inv);
  }
}

extern "C" void kernel_launch(void* const* d_in, const int* in_sizes, int n_in,
                              void* d_out, int out_size, void* d_ws, size_t ws_size,
                              hipStream_t stream) {
  (void)in_sizes; (void)n_in; (void)out_size; (void)ws_size;
  const float* x     = (const float*)d_in[0];   // [B,T,C]
  const float* w_qkv = (const float*)d_in[1];   // [C, 3C]
  const float* w_out = (const float*)d_in[2];   // [C, C]
  const float* b_out = (const float*)d_in[3];   // [C]
  float* out = (float*)d_out;

  bf16* xbf   = (bf16*)d_ws;                       // 8192*1024
  bf16* wqkvT = xbf   + (size_t)BT * CDIM;         // 3072*1024
  bf16* woutT = wqkvT + (size_t)C3 * CDIM;         // 1024*1024
  bf16* qkv   = woutT + (size_t)CDIM * CDIM;       // 8192*3072
  bf16* ctx   = qkv   + (size_t)BT * C3;           // 8192*1024

  // 1. convert / transpose inputs to bf16
  {
    int n4 = BT * CDIM / 4;
    cvt_f32_bf16<<<(n4 + 255) / 256, 256, 0, stream>>>(x, xbf, n4);
  }
  cvt_transpose<<<(C3 * CDIM + 255) / 256, 256, 0, stream>>>(w_qkv, wqkvT, CDIM, C3);
  cvt_transpose<<<(CDIM * CDIM + 255) / 256, 256, 0, stream>>>(w_out, woutT, CDIM, CDIM);

  // 2. QKV projection: [8192,1024] @ [1024,3072] -> bf16 [8192,3072]
  gemm_bf16_wmma<1, 0><<<dim3(C3 / 128, BT / 128), 256, 0, stream>>>(
      xbf, wqkvT, nullptr, qkv, BT, C3, CDIM);

  // 3. causal flash attention -> bf16 ctx [8192,1024]
  attn_flash_wmma<<<dim3(TSEQ / 64, 4 * NHEAD), 128, 0, stream>>>(qkv, ctx);

  // 4. output projection + bias: [8192,1024] @ [1024,1024] -> f32 out
  gemm_bf16_wmma<0, 1><<<dim3(CDIM / 128, BT / 128), 256, 0, stream>>>(
      ctx, woutT, b_out, out, BT, CDIM, CDIM);
}
